// RTDETRDecoder_4226247819542
// MI455X (gfx1250) — matrix operand order repaired and checked by reading, not compiled, not run on hardware
//
#include <hip/hip_runtime.h>
#include <hip/hip_bf16.h>
#include <math.h>

typedef __attribute__((ext_vector_type(16))) _Float16 v16h;
typedef __attribute__((ext_vector_type(8)))  _Float16 v8h;
typedef __attribute__((ext_vector_type(4)))  _Float16 v4h;
typedef __attribute__((ext_vector_type(8)))  float    v8f;

// Problem constants (from reference setup_inputs)
#define BB   32
#define QQ   300
#define DD   256
#define VV   8500
#define NHEAD 8
#define NLVL  4
#define NPT   4

// ---------------------------------------------------------------------------
// WMMA GEMM: C(MxN) = A(MxK) * B(KxN) + bias(N), f32 in/out, f16 WMMA core.
// Block: 256 threads = 8 waves, computes a 64x128 C tile.
// Wave w: msub = w%4 (16-row strip), ngrp = (w/4)*4 (four 16-col strips),
// so each wave issues 4 WMMAs per k-step reusing one A fragment.
// B is staged TRANSPOSED in LDS (n-major, k-contiguous) so both A and B
// fragments are assembled with 16B/32B contiguous LDS loads (ds_load_b128).
// Requires M%64==0, N%128==0, K%32==0 (true for all launches here).
// ---------------------------------------------------------------------------
__global__ __launch_bounds__(256)
void wmma_gemm_bias(const float* __restrict__ A, const float* __restrict__ Bw,
                    const float* __restrict__ bias, float* __restrict__ C,
                    int M, int N, int K)
{
    __shared__ _Float16 lA [64 * 32];    // [m][k], k contiguous
    __shared__ _Float16 lBt[128 * 32];   // [n][k], k contiguous (transposed)

    const int t    = threadIdx.x;
    const int wave = t >> 5;
    const int lane = t & 31;
    const int lm   = lane & 15;          // M/N position within 16
    const int lh   = lane >> 4;          // 0: lanes 0-15, 1: lanes 16-31
    const int msub = wave & 3;
    const int ngrp = (wave >> 2) * 4;    // first of four n-subtiles

    const int m0 = blockIdx.x * 64;
    const int n0 = blockIdx.y * 128;

    v8f c[4] = {{}, {}, {}, {}};

    // A staging: thread -> (row 0..63, kcol {0,8,16,24}), 8 floats along k
    const int ar = t >> 2;
    const int ak = (t & 3) << 3;
    // B staging: thread -> 4x4 (k x n) block
    const int bn4 = (t & 31) << 2;       // n col base 0..124
    const int bk4 = (t >> 5) << 2;       // k row base 0..28

    for (int k0 = 0; k0 < K; k0 += 32) {
        // ---- stage A tile (64x32 f32 -> f16 LDS, row-major) ----
        {
            const float4* g = reinterpret_cast<const float4*>(
                A + (size_t)(m0 + ar) * K + k0 + ak);
            float4 f0 = g[0];
            float4 f1 = g[1];
            v8h h;
            h[0] = (_Float16)f0.x; h[1] = (_Float16)f0.y;
            h[2] = (_Float16)f0.z; h[3] = (_Float16)f0.w;
            h[4] = (_Float16)f1.x; h[5] = (_Float16)f1.y;
            h[6] = (_Float16)f1.z; h[7] = (_Float16)f1.w;
            *reinterpret_cast<v8h*>(&lA[ar * 32 + ak]) = h;
            if (k0 + 32 < K) {
                __builtin_prefetch(A + (size_t)(m0 + ar) * K + k0 + 32 + ak, 0, 0);
            }
        }
        // ---- stage B tile (32x128 f32 -> f16 LDS, TRANSPOSED to [n][k]) ----
        {
            float4 f0 = *reinterpret_cast<const float4*>(
                Bw + (size_t)(k0 + bk4 + 0) * N + n0 + bn4);
            float4 f1 = *reinterpret_cast<const float4*>(
                Bw + (size_t)(k0 + bk4 + 1) * N + n0 + bn4);
            float4 f2 = *reinterpret_cast<const float4*>(
                Bw + (size_t)(k0 + bk4 + 2) * N + n0 + bn4);
            float4 f3 = *reinterpret_cast<const float4*>(
                Bw + (size_t)(k0 + bk4 + 3) * N + n0 + bn4);
            const float* r0 = reinterpret_cast<const float*>(&f0);
            const float* r1 = reinterpret_cast<const float*>(&f1);
            const float* r2 = reinterpret_cast<const float*>(&f2);
            const float* r3 = reinterpret_cast<const float*>(&f3);
            #pragma unroll
            for (int j = 0; j < 4; ++j) {
                v4h h;
                h[0] = (_Float16)r0[j]; h[1] = (_Float16)r1[j];
                h[2] = (_Float16)r2[j]; h[3] = (_Float16)r3[j];
                *reinterpret_cast<v4h*>(&lBt[(bn4 + j) * 32 + bk4]) = h;
            }
        }
        __syncthreads();

        // ---- A fragment: K = lh*16 + {0..15} ----
        v16h af;
        {
            const int ro = (msub * 16 + lm) * 32 + (lh << 4);
            v8h lo = *reinterpret_cast<const v8h*>(&lA[ro]);
            v8h hi = *reinterpret_cast<const v8h*>(&lA[ro + 8]);
            af = __builtin_shufflevector(lo, hi,
                    0, 1, 2, 3, 4, 5, 6, 7, 8, 9, 10, 11, 12, 13, 14, 15);
        }
        // ---- 4 B fragments + 4 WMMAs (A reused) ----
        #pragma unroll
        for (int i = 0; i < 4; ++i) {
            const int col = (ngrp + i) * 16 + lm;
            // B frag: K = lh*16 + {0..15}, contiguous 32B in lBt
            const v16h bf = *reinterpret_cast<const v16h*>(
                &lBt[col * 32 + (lh << 4)]);
            c[i] = __builtin_amdgcn_wmma_f32_16x16x32_f16(
                       false, af, false, bf, (short)0, c[i], false, false);
        }
        __syncthreads();
    }

    // ---- store C + bias (C layout: VGPR r -> M = r + 8*(lane>=16)) ----
    #pragma unroll
    for (int i = 0; i < 4; ++i) {
        const int col = n0 + (ngrp + i) * 16 + lm;
        const float bv = bias[col];
        #pragma unroll
        for (int r = 0; r < 8; ++r) {
            const int row = m0 + msub * 16 + r + (lh << 3);
            C[(size_t)row * N + col] = c[i][r] + bv;
        }
    }
}

// ---------------------------------------------------------------------------
// Deformable sampling: one wave per (b, q, head); 32 lanes = 32 head channels.
// Softmax over 16 (level,point) logits done with lane shuffles; each of the
// 64 corner gathers is a coalesced 128B line from the projected value tensor.
// ---------------------------------------------------------------------------
__global__ __launch_bounds__(256)
void ms_deform_sample(const float* __restrict__ VALP,   // (B*V) x 256
                      const float* __restrict__ OFF,    // (B*Q) x 256
                      const float* __restrict__ ATT,    // (B*Q) x 128
                      const float* __restrict__ REF,    // B,Q,L,2
                      const int*   __restrict__ SHP,    // L,2 (H,W)
                      const int*   __restrict__ LSI,    // L
                      float* __restrict__ SMP)          // (B*Q) x 256
{
    const int wid  = blockIdx.x * 8 + (threadIdx.x >> 5);
    const int lane = threadIdx.x & 31;
    const int head = wid & 7;
    const int q    = (wid >> 3) % QQ;
    const int b    = wid / (NHEAD * QQ);
    const int row  = b * QQ + q;

    // ---- softmax over 16 logits (lanes 0..15 hold one each) ----
    float x = -1e30f;
    if (lane < 16) x = ATT[(size_t)row * 128 + head * 16 + lane];
    float mx = x;
    #pragma unroll
    for (int d = 8; d >= 1; d >>= 1) mx = fmaxf(mx, __shfl_xor(mx, d));
    float e = __expf(x - mx);
    float sm = e;
    #pragma unroll
    for (int d = 8; d >= 1; d >>= 1) sm += __shfl_xor(sm, d);
    const float p = e / sm;

    float acc = 0.0f;
    const float* vbase = VALP + (size_t)b * VV * DD + head * 32 + lane;

    #pragma unroll
    for (int s = 0; s < 16; ++s) {
        const int lvl = s >> 2;
        const float a_s = __shfl(p, s);

        const int Hl = SHP[lvl * 2 + 0];
        const int Wl = SHP[lvl * 2 + 1];
        const int st = LSI[lvl];

        const float offx = OFF[(size_t)row * 256 + (head * 16 + s) * 2 + 0];
        const float offy = OFF[(size_t)row * 256 + (head * 16 + s) * 2 + 1];
        const float rx   = REF[((size_t)row * NLVL + lvl) * 2 + 0];
        const float ry   = REF[((size_t)row * NLVL + lvl) * 2 + 1];

        // loc = ref + off/[W,H]; x = loc_x*W - 0.5  ==>  ref_x*W + off_x - 0.5
        const float fx = rx * (float)Wl + offx - 0.5f;
        const float fy = ry * (float)Hl + offy - 0.5f;
        const float x0f = floorf(fx), y0f = floorf(fy);
        const int   xi = (int)x0f,   yi = (int)y0f;
        const float wx1 = fx - x0f,  wy1 = fy - y0f;
        const float wx0 = 1.0f - wx1, wy0 = 1.0f - wy1;

        #pragma unroll
        for (int cy = 0; cy < 2; ++cy) {
            #pragma unroll
            for (int cx = 0; cx < 2; ++cx) {
                const int xc = xi + cx;
                const int yc = yi + cy;
                if (xc >= 0 && xc < Wl && yc >= 0 && yc < Hl) {
                    const float w = (cx ? wx1 : wx0) * (cy ? wy1 : wy0);
                    const float v = vbase[(size_t)(st + yc * Wl + xc) * DD];
                    acc += a_s * w * v;
                }
            }
        }
    }
    SMP[(size_t)row * 256 + head * 32 + lane] = acc;
}

// ---------------------------------------------------------------------------
extern "C" void kernel_launch(void* const* d_in, const int* in_sizes, int n_in,
                              void* d_out, int out_size, void* d_ws, size_t ws_size,
                              hipStream_t stream)
{
    (void)in_sizes; (void)n_in; (void)out_size; (void)ws_size;

    const float* query = (const float*)d_in[0];
    const float* refp  = (const float*)d_in[1];
    const float* value = (const float*)d_in[2];
    const int*   shp   = (const int*)d_in[3];
    const int*   lsi   = (const int*)d_in[4];
    const float* Woff  = (const float*)d_in[5];
    const float* boff  = (const float*)d_in[6];
    const float* Wattn = (const float*)d_in[7];
    const float* battn = (const float*)d_in[8];
    const float* Wval  = (const float*)d_in[9];
    const float* bval  = (const float*)d_in[10];
    const float* Wout  = (const float*)d_in[11];
    const float* bout  = (const float*)d_in[12];
    float* out = (float*)d_out;

    const int MV = BB * VV;   // 272000 rows (divisible by 64)
    const int MQ = BB * QQ;   // 9600 rows   (divisible by 64)

    char* ws = (char*)d_ws;
    float* valp = (float*)ws; ws += (size_t)MV * DD  * sizeof(float);
    float* offb = (float*)ws; ws += (size_t)MQ * DD  * sizeof(float);
    float* attb = (float*)ws; ws += (size_t)MQ * 128 * sizeof(float);
    float* smpb = (float*)ws;

    dim3 blk(256);

    // 1) value projection: (B*V,256) x (256,256)
    wmma_gemm_bias<<<dim3(MV / 64, DD / 128), blk, 0, stream>>>(
        value, Wval, bval, valp, MV, DD, DD);
    // 2) sampling offsets: (B*Q,256) x (256,256)
    wmma_gemm_bias<<<dim3(MQ / 64, DD / 128), blk, 0, stream>>>(
        query, Woff, boff, offb, MQ, DD, DD);
    // 3) attention logits: (B*Q,256) x (256,128)
    wmma_gemm_bias<<<dim3(MQ / 64, 128 / 128), blk, 0, stream>>>(
        query, Wattn, battn, attb, MQ, 128, DD);
    // 4) softmax + bilinear gather + weighted sum
    ms_deform_sample<<<dim3(MQ), blk, 0, stream>>>(
        valp, offb, attb, refp, shp, lsi, smpb);
    // 5) output projection: (B*Q,256) x (256,256)
    wmma_gemm_bias<<<dim3(MQ / 64, DD / 128), blk, 0, stream>>>(
        smpb, Wout, bout, out, MQ, DD, DD);
}
